// B4Block_63385127355014
// MI455X (gfx1250) — matrix operand order repaired and checked by reading, not compiled, not run on hardware
//
#include <hip/hip_runtime.h>

// ---------------------------------------------------------------------------
// Types / helpers
// ---------------------------------------------------------------------------
typedef __attribute__((ext_vector_type(16))) __bf16 v16bf;
typedef __attribute__((ext_vector_type(8)))  float  v8f;

__device__ __forceinline__ float bf2f(unsigned short h) {
    union { unsigned u; float f; } x; x.u = ((unsigned)h) << 16; return x.f;
}
__device__ __forceinline__ unsigned short f2bf(float f) {
    union { float f; unsigned u; } x; x.f = f;
    unsigned u = x.u;
    unsigned r = (u + 0x7FFFu + ((u >> 16) & 1u)) >> 16;
    return (unsigned short)r;
}
__device__ __forceinline__ float siluf(float x)  { return x / (1.f + __expf(-x)); }
__device__ __forceinline__ float sigm(float x)   { return 1.f / (1.f + __expf(-x)); }
__device__ __forceinline__ float hswish(float x) {
    float r = x + 3.f; r = fminf(fmaxf(r, 0.f), 6.f); return x * r * (1.f / 6.f);
}

// A fragment (16x32 bf16, contiguous k order): p = wrow + k0 (wrow pre-offset by half*16)
__device__ __forceinline__ v16bf loadA(const unsigned short* p) {
    union { v16bf v; uint4 q[2]; } u;
    u.q[0] = *(const uint4*)(p);        // K: k0+hb .. +7
    u.q[1] = *(const uint4*)(p + 8);    // K: k0+hb+8 .. +7
    return u.v;
}

// B fragment (32x16 bf16, contiguous k order): p = brow + k0 (brow pre-offset by half*16)
template <bool AFF, int INACT>
__device__ __forceinline__ v16bf loadB(const unsigned short* p,
                                       const float* __restrict__ isc,
                                       const float* __restrict__ ish, int cb) {
    union { v16bf v; uint4 q[2]; unsigned short us[16]; } u;
    u.q[0] = *(const uint4*)(p);
    u.q[1] = *(const uint4*)(p + 8);
    if (AFF) {
#pragma unroll
        for (int j = 0; j < 16; ++j) {
            float x = bf2f(u.us[j]);
            x = x * isc[cb + j] + ish[cb + j];
            if (INACT == 1)      x = siluf(x);
            else if (INACT == 2) x = hswish(x);
            u.us[j] = f2bf(x);
        }
    }
    return u.v;
}

// ---------------------------------------------------------------------------
// k_gemm<K,AFF,INACT,EPI>: out[n][m] = act[remap(n)][k] . W[m][k] + bias[m]
// bf16 in (channels-last rows), f32 WMMA accumulate, bf16 out.
// K loop: outer no-unroll over 96-wide chunks (bounds live registers ->
// multiple waves/SIMD), inner fully unrolled 3 k-steps x 4 WMMAs on 4
// independent accumulators. Branch-free around every WMMA (EXEC all-ones).
// Next chunk is prefetched (global_prefetch) at chunk start.
// ---------------------------------------------------------------------------
template <int K, bool AFF, int INACT, int EPI>
__global__ __launch_bounds__(256) void k_gemm(
    const unsigned short* __restrict__ act, const unsigned short* __restrict__ W,
    const float* __restrict__ bias, unsigned short* __restrict__ out,
    long long N, int M,
    int Vv, int Tout, int Tin, int s, int toff,
    const float* __restrict__ isc, const float* __restrict__ ish)
{
    static_assert(K % 96 == 0, "K must be a multiple of 96");
    const int lane = threadIdx.x & 31;
    const int wid  = blockIdx.x * 8 + (threadIdx.x >> 5);
    const int mT   = M >> 4;
    const int mt   = wid % mT;
    const long long nbase = (long long)(wid / mT) * 64;
    if (nbase >= N) return;                       // wave-uniform
    const int half = lane >> 4;                   // 0|1
    const int l16  = lane & 15;

    const unsigned short* wrow = W + (size_t)(mt * 16 + l16) * K + half * 16;

    // B row pointers, clamped so every load is legal (dead subtiles recompute
    // row N-1; their stores are skipped). No guards inside the main loop.
    const unsigned short* brow[4];
#pragma unroll
    for (int u = 0; u < 4; ++u) {
        long long n = nbase + u * 16 + l16;
        if (n >= N) n = N - 1;
        long long v = n % Vv;
        long long t = (n / Vv) % Tout;
        long long b = n / ((long long)Vv * Tout);
        long long ir = (b * (long long)Tin + t * (long long)s + toff) * Vv + v;
        brow[u] = act + (size_t)ir * K + half * 16;
    }
    const unsigned short* b0 = brow[0];
    const unsigned short* b1 = brow[1];
    const unsigned short* b2 = brow[2];
    const unsigned short* b3 = brow[3];

    v8f a0 = {}, a1 = {}, a2 = {}, a3 = {};

#pragma unroll 1
    for (int kc = 0; kc < K; kc += 96) {
        if (kc + 96 < K) {                        // prefetch next chunk
            __builtin_prefetch(wrow + kc + 96, 0, 1);
            __builtin_prefetch(b0 + kc + 96, 0, 1);
            __builtin_prefetch(b1 + kc + 96, 0, 1);
            __builtin_prefetch(b2 + kc + 96, 0, 1);
            __builtin_prefetch(b3 + kc + 96, 0, 1);
        }
#pragma unroll
        for (int kk = 0; kk < 96; kk += 32) {
            const int k0 = kc + kk;
            const int cb = k0 + half * 16;
            v16bf Af = loadA(wrow + k0);
            v16bf B0 = loadB<AFF, INACT>(b0 + k0, isc, ish, cb);
            v16bf B1 = loadB<AFF, INACT>(b1 + k0, isc, ish, cb);
            v16bf B2 = loadB<AFF, INACT>(b2 + k0, isc, ish, cb);
            v16bf B3 = loadB<AFF, INACT>(b3 + k0, isc, ish, cb);
            a0 = __builtin_amdgcn_wmma_f32_16x16x32_bf16(false, Af, false, B0, (short)0, a0, false, false);
            a1 = __builtin_amdgcn_wmma_f32_16x16x32_bf16(false, Af, false, B1, (short)0, a1, false, false);
            a2 = __builtin_amdgcn_wmma_f32_16x16x32_bf16(false, Af, false, B2, (short)0, a2, false, false);
            a3 = __builtin_amdgcn_wmma_f32_16x16x32_bf16(false, Af, false, B3, (short)0, a3, false, false);
        }
    }

    const int mb = mt * 16 + half * 8;
    float bv[8];
#pragma unroll
    for (int r = 0; r < 8; ++r) bv[r] = bias ? bias[mb + r] : 0.f;

    v8f accs[4] = { a0, a1, a2, a3 };
#pragma unroll
    for (int u = 0; u < 4; ++u) {
        if ((nbase + u * 16) >= N) continue;      // wave-uniform store guard
        long long n = nbase + u * 16 + l16;
        unsigned short o8[8];
#pragma unroll
        for (int r = 0; r < 8; ++r) {
            float vv = accs[u][r] + bv[r];
            if (EPI == 1) vv = sigm(vv);
            o8[r] = f2bf(vv);
        }
        *(uint4*)(out + (size_t)n * M + mb) = *(const uint4*)o8;
    }
}

// ---------------------------------------------------------------------------
// Adjacency matmul + NCHW->NTVC transpose. z[n][k*96+c] = sum_u x[b,c,t,u]*A[k][u][v]
// A-hat staged in LDS. block = 96 threads (c), grid = B*T.
// ---------------------------------------------------------------------------
__global__ void k_adj(const float* __restrict__ x, const float* __restrict__ Ah,
                      unsigned short* __restrict__ z, unsigned short* __restrict__ xb,
                      int Bn, int Cin, int T, int Vv, int Kk)
{
    __shared__ float As[3 * 25 * 25];
    for (int i = threadIdx.x; i < Kk * Vv * Vv; i += blockDim.x) As[i] = Ah[i];
    __syncthreads();
    const int c = threadIdx.x;
    const int bt = blockIdx.x;
    const int t = bt % T, b = bt / T;
    float xr[25];
    const float* xp = x + (((size_t)b * Cin + c) * T + t) * Vv;
#pragma unroll
    for (int v = 0; v < 25; ++v) xr[v] = xp[v];
    const long long nb = ((long long)b * T + t) * Vv;
    for (int v = 0; v < 25; ++v) xb[(nb + v) * Cin + c] = f2bf(xr[v]);
    for (int k = 0; k < 3; ++k) {
        for (int v = 0; v < 25; ++v) {
            float sAcc = 0.f;
#pragma unroll
            for (int u = 0; u < 25; ++u) sAcc += xr[u] * As[(k * 25 + u) * 25 + v];
            z[(nb + v) * (size_t)(Kk * Cin) + k * Cin + c] = f2bf(sAcc);
        }
    }
}

// ---------------------------------------------------------------------------
// Depthwise (5,1) temporal conv, channels-last, optional fused input BN(+SiLU).
// ---------------------------------------------------------------------------
__global__ void k_dwconv(const unsigned short* __restrict__ in, unsigned short* __restrict__ out,
                         const float* __restrict__ w, const float* __restrict__ bias,
                         const float* __restrict__ isc, const float* __restrict__ ish, int inAct,
                         int Bn, int Tin, int Tout, int Vv, int C, int s)
{
    long long idx = (long long)blockIdx.x * blockDim.x + threadIdx.x;
    long long tot = (long long)Bn * Tout * Vv * C;
    if (idx >= tot) return;
    int c = (int)(idx % C);
    long long r = idx / C;
    int v = (int)(r % Vv); r /= Vv;
    int t = (int)(r % Tout);
    int b = (int)(r / Tout);
    float acc = bias[c];
#pragma unroll
    for (int i = 0; i < 5; ++i) {
        int ti = t * s + i - 2;
        if (ti < 0 || ti >= Tin) continue;
        float xv = bf2f(in[(((long long)b * Tin + ti) * Vv + v) * C + c]);
        if (isc) { xv = xv * isc[c] + ish[c]; if (inAct == 1) xv = siluf(xv); }
        acc += w[c * 5 + i] * xv;
    }
    out[idx] = f2bf(acc);
}

// ---------------------------------------------------------------------------
// Per-channel sum / sumsq (training BN stats). block = C threads, f32 atomics.
// ---------------------------------------------------------------------------
__global__ void k_stats(const unsigned short* __restrict__ x, float* __restrict__ st,
                        int C, long long rows)
{
    const int c = threadIdx.x;
    float s0 = 0.f, s1 = 0.f;
    for (long long r = blockIdx.x; r < rows; r += gridDim.x) {
        float v = bf2f(x[r * C + c]);
        s0 += v; s1 += v * v;
    }
    atomicAdd(&st[c], s0);
    atomicAdd(&st[C + c], s1);
}

__global__ void k_bnfin(const float* __restrict__ st, const float* __restrict__ g,
                        const float* __restrict__ b, float* __restrict__ sc,
                        float* __restrict__ sh, float invN, int C)
{
    int c = threadIdx.x;
    if (c >= C) return;
    float m = st[c] * invN;
    float var = st[C + c] * invN - m * m;
    float s = g[c] * rsqrtf(var + 1e-5f);
    sc[c] = s;
    sh[c] = b[c] - m * s;
}

// ---------------------------------------------------------------------------
// Elementwise combiners
// ---------------------------------------------------------------------------
__global__ void k_comb_gcn(const unsigned short* __restrict__ ga, const unsigned short* __restrict__ rr,
                           const float* scA, const float* shA, const float* scB, const float* shB,
                           const float* __restrict__ je, unsigned short* __restrict__ out,
                           long long rows, int C, int Vv)
{
    long long idx = (long long)blockIdx.x * blockDim.x + threadIdx.x;
    if (idx >= rows * C) return;
    int c = (int)(idx % C);
    long long n = idx / C;
    int v = (int)(n % Vv);
    float a = bf2f(ga[idx]) * scA[c] + shA[c];
    float b = bf2f(rr[idx]) * scB[c] + shB[c];
    out[idx] = f2bf(siluf(a + b) + je[c * Vv + v]);
}

__global__ void k_comb_tcn(const unsigned short* __restrict__ yr, const unsigned short* __restrict__ rr,
                           const float* scA, const float* shA, const float* scB, const float* shB,
                           int resBN, unsigned short* __restrict__ out, long long total, int C)
{
    long long idx = (long long)blockIdx.x * blockDim.x + threadIdx.x;
    if (idx >= total) return;
    int c = (int)(idx % C);
    float a = bf2f(yr[idx]) * scA[c] + shA[c];
    float r = bf2f(rr[idx]);
    if (resBN) r = r * scB[c] + shB[c];
    out[idx] = f2bf(a + r);
}

// attention mean-pool: rows b*(T+V)+j; j<T -> mean over V, else mean over T
__global__ void k_pool(const unsigned short* __restrict__ x, unsigned short* __restrict__ out,
                       int Bn, int T, int Vv, int C)
{
    int row = blockIdx.x, c = threadIdx.x;
    int b = row / (T + Vv), j = row % (T + Vv);
    float s = 0.f;
    if (j < T) {
        for (int v = 0; v < Vv; ++v) s += bf2f(x[(((long long)b * T + j) * Vv + v) * C + c]);
        s *= (1.f / (float)Vv);
    } else {
        int v = j - T;
        for (int t = 0; t < T; ++t) s += bf2f(x[(((long long)b * T + t) * Vv + v) * C + c]);
        s *= (1.f / (float)T);
    }
    out[(long long)row * C + c] = f2bf(s);
}

__global__ void k_mul(const unsigned short* __restrict__ x, const unsigned short* __restrict__ at,
                      const unsigned short* __restrict__ av, unsigned short* __restrict__ out,
                      int Bn, int T, int Vv, int C)
{
    long long idx = (long long)blockIdx.x * blockDim.x + threadIdx.x;
    long long tot = (long long)Bn * T * Vv * C;
    if (idx >= tot) return;
    int c = (int)(idx % C);
    long long n = idx / C;
    int v = (int)(n % Vv);
    int t = (int)((n / Vv) % T);
    int b = (int)(n / ((long long)Vv * T));
    float val = bf2f(x[idx]) * bf2f(at[((long long)b * T + t) * C + c])
                             * bf2f(av[((long long)b * Vv + v) * C + c]);
    out[idx] = f2bf(val);
}

// final: silu(bn(m) + x3), write fp32 NCHW
__global__ void k_final(const unsigned short* __restrict__ m, const unsigned short* __restrict__ x,
                        const float* __restrict__ sc, const float* __restrict__ sh,
                        float* __restrict__ out, int Bn, int T, int Vv, int C)
{
    long long idx = (long long)blockIdx.x * blockDim.x + threadIdx.x;
    long long tot = (long long)Bn * T * Vv * C;
    if (idx >= tot) return;
    int c = (int)(idx % C);
    long long n = idx / C;
    int v = (int)(n % Vv);
    int t = (int)((n / Vv) % T);
    int b = (int)(n / ((long long)Vv * T));
    float y = siluf(bf2f(m[idx]) * sc[c] + sh[c] + bf2f(x[idx]));
    out[(((long long)b * C + c) * T + t) * Vv + v] = y;
}

// ---------------------------------------------------------------------------
// Weight prep
// ---------------------------------------------------------------------------
__global__ void k_cvt(const float* __restrict__ s, unsigned short* __restrict__ d, int n) {
    int i = blockIdx.x * blockDim.x + threadIdx.x;
    if (i < n) d[i] = f2bf(s[i]);
}
__global__ void k_prep_wg(const float* __restrict__ gw, unsigned short* __restrict__ d) {
    int i = blockIdx.x * blockDim.x + threadIdx.x;           // [192][288]
    if (i >= 192 * 288) return;
    int o = i / 288, kc = i % 288, k = kc / 96, c = kc % 96;
    d[i] = f2bf(gw[((size_t)k * 192 + o) * 96 + c]);
}
__global__ void k_sum_bias(const float* __restrict__ gb, float* __restrict__ d) {
    int o = threadIdx.x;
    d[o] = gb[o] + gb[192 + o] + gb[384 + o];
}
__global__ void k_prep_adj(const float* __restrict__ Af, const float* __restrict__ Ae,
                           float* __restrict__ d, int n) {
    int i = blockIdx.x * blockDim.x + threadIdx.x;
    if (i < n) d[i] = Af[i] * Ae[i];
}

// ---------------------------------------------------------------------------
// Host launch
// ---------------------------------------------------------------------------
#define LGEMM(Kc, AFFc, IAc, EPc, act, W, bias, out, Nn, Mm, Vv, To, Ti, ss, toff, iscp, ishp)      \
    do {                                                                                            \
        long long waves_ = (long long)((Mm) / 16) * (((Nn) + 63) / 64);                             \
        int blocks_ = (int)((waves_ + 7) / 8);                                                      \
        k_gemm<Kc, AFFc, IAc, EPc><<<blocks_, 256, 0, stream>>>(                                    \
            act, W, bias, out, (long long)(Nn), Mm, Vv, To, Ti, ss, toff, iscp, ishp);              \
    } while (0)

extern "C" void kernel_launch(void* const* d_in, const int* in_sizes, int n_in,
                              void* d_out, int out_size, void* d_ws, size_t ws_size,
                              hipStream_t stream)
{
    (void)in_sizes; (void)n_in; (void)out_size; (void)ws_size;
    // jax pytree flatten order (dict keys sorted): A_fixed, params..., x
    enum {
        I_AFIX = 0, I_AEDGE = 1,
        I_ATT_BN_B = 2, I_ATT_BN_G = 3, I_ATT_CT_B = 4, I_ATT_CT_W = 5,
        I_ATT_CV_B = 6, I_ATT_CV_W = 7, I_ATT_FCN_B = 8, I_ATT_FCN_BN_B = 9,
        I_ATT_FCN_BN_G = 10, I_ATT_FCN_W = 11,
        I_GCN_B = 12, I_GCN_BN_B = 13, I_GCN_BN_G = 14, I_GCN_W = 15, I_JE = 16,
        I_RES_B = 17, I_RES_BN_B = 18, I_RES_BN_G = 19, I_RES_W = 20,
        T0 = 21, T1 = 41, I_X = 57
    };
    // per-tcn sorted offsets: bn1_b..bn4_g = 0..7, dw1_b=8, dw1_w=9, dw2_b=10,
    // dw2_w=11, pw1_b=12, pw1_w=13, pw2_b=14, pw2_w=15, res_b=16..res_w=19
    auto IN = [&](int i) { return (const float*)d_in[i]; };

    const long long N1 = 32LL * 288 * 25;   // 230400
    const long long N2 = 32LL * 144 * 25;   // 115200

    char* bw = (char*)d_ws;
    const size_t OFF_STATS = 0;
    const size_t OFF_SS    = 64 * 1024;
    const size_t OFF_ADJ   = 128 * 1024;
    const size_t OFF_BIASG = 160 * 1024;
    const size_t OFF_WG    = 192 * 1024;
    const size_t OFF_WRES  = OFF_WG    + 131072;
    const size_t OFF_PW1_0 = OFF_WRES  + 65536;
    const size_t OFF_PW2_0 = OFF_PW1_0 + 65536;
    const size_t OFF_PW1_1 = OFF_PW2_0 + 65536;
    const size_t OFF_PW2_1 = OFF_PW1_1 + 65536;
    const size_t OFF_RESW  = OFF_PW2_1 + 65536;
    const size_t OFF_FCNW  = OFF_RESW  + 131072;
    const size_t OFF_CTW   = OFF_FCNW  + 65536;
    const size_t OFF_CVW   = OFF_CTW   + 65536;
    const size_t ACT0      = 1048576;
    const size_t A_XB = ACT0;
    const size_t A_Z  = A_XB + (size_t)N1 * 96 * 2;
    const size_t A_S0 = A_Z  + (size_t)N1 * 288 * 2;
    const size_t A_S1 = A_S0 + (size_t)N1 * 192 * 2;
    const size_t A_S2 = A_S1 + (size_t)N1 * 192 * 2;
    const size_t A_S3 = A_S2 + (size_t)N1 * 192 * 2;
    const size_t A_P  = A_S3 + (size_t)N2 * 192 * 2;
    const size_t A_FC = A_P  + 5408 * 192 * 2;
    const size_t A_AT = A_FC + 5408 * 96 * 2;
    const size_t A_AV = A_AT + 4608 * 192 * 2;

    auto F = [&](size_t o) { return (float*)(bw + o); };
    auto H = [&](size_t o) { return (unsigned short*)(bw + o); };
    auto SC = [&](int s) { return F(OFF_SS) + s * 384; };
    auto SH = [&](int s) { return F(OFF_SS) + s * 384 + 192; };
    auto ST = [&](int s) { return F(OFF_STATS) + s * 384; };

    auto stats = [&](const unsigned short* x, int slot, int C, long long rows) {
        k_stats<<<512, C, 0, stream>>>(x, ST(slot), C, rows);
    };
    auto bnfin = [&](int slot, const float* g, const float* b, float invN, int C) {
        k_bnfin<<<1, C, 0, stream>>>(ST(slot), g, b, SC(slot), SH(slot), invN, C);
    };
    const float inv1 = 1.f / (float)N1, inv2 = 1.f / (float)N2, invP = 1.f / 5408.f;
    const float* nilf = nullptr;

    // 0. zero BN stats arena (re-done every call -> deterministic)
    hipMemsetAsync(bw + OFF_STATS, 0, 13 * 384 * sizeof(float), stream);

    // 1. weight prep (fp32 -> bf16, permutes, fused adjacency, summed gcn bias)
    k_prep_adj<<<(1875 + 255) / 256, 256, 0, stream>>>(IN(I_AFIX), IN(I_AEDGE), F(OFF_ADJ), 1875);
    k_prep_wg<<<(55296 + 255) / 256, 256, 0, stream>>>(IN(I_GCN_W), H(OFF_WG));
    k_sum_bias<<<1, 192, 0, stream>>>(IN(I_GCN_B), F(OFF_BIASG));
    k_cvt<<<72, 256, 0, stream>>>(IN(I_RES_W),     H(OFF_WRES),  192 * 96);
    k_cvt<<<72, 256, 0, stream>>>(IN(T0 + 13),     H(OFF_PW1_0),  96 * 192);
    k_cvt<<<72, 256, 0, stream>>>(IN(T0 + 15),     H(OFF_PW2_0), 192 * 96);
    k_cvt<<<72, 256, 0, stream>>>(IN(T1 + 13),     H(OFF_PW1_1),  96 * 192);
    k_cvt<<<72, 256, 0, stream>>>(IN(T1 + 15),     H(OFF_PW2_1), 192 * 96);
    k_cvt<<<144, 256, 0, stream>>>(IN(T0 + 19),    H(OFF_RESW),  192 * 192);
    k_cvt<<<72, 256, 0, stream>>>(IN(I_ATT_FCN_W), H(OFF_FCNW),   96 * 192);
    k_cvt<<<72, 256, 0, stream>>>(IN(I_ATT_CT_W),  H(OFF_CTW),   192 * 96);
    k_cvt<<<72, 256, 0, stream>>>(IN(I_ATT_CV_W),  H(OFF_CVW),   192 * 96);

    // 2. adjacency matmuls + transpose to channels-last (z: 288ch, xb: 96ch)
    k_adj<<<32 * 288, 96, 0, stream>>>(IN(I_X), F(OFF_ADJ), H(A_Z), H(A_XB), 32, 96, 288, 25, 3);

    // 3. GCN GEMMs (WMMA) + BN stats + combine -> x1 @ S2
    LGEMM(288, false, 0, 0, H(A_Z),  H(OFF_WG),   F(OFF_BIASG), H(A_S0), N1, 192, 25, 288, 288, 1, 0, nilf, nilf);
    LGEMM( 96, false, 0, 0, H(A_XB), H(OFF_WRES), IN(I_RES_B),  H(A_S1), N1, 192, 25, 288, 288, 1, 0, nilf, nilf);
    stats(H(A_S0), 0, 192, N1); stats(H(A_S1), 1, 192, N1);
    bnfin(0, IN(I_GCN_BN_G), IN(I_GCN_BN_B), inv1, 192);
    bnfin(1, IN(I_RES_BN_G), IN(I_RES_BN_B), inv1, 192);
    {
        long long tot = N1 * 192;
        k_comb_gcn<<<(int)((tot + 255) / 256), 256, 0, stream>>>(
            H(A_S0), H(A_S1), SC(0), SH(0), SC(1), SH(1), IN(I_JE), H(A_S2), N1, 192, 25);
    }

    // 4. TCN layer 0 (stride 2), input x1 @ S2
    {
        long long tot = N1 * 192;
        k_dwconv<<<(int)((tot + 255) / 256), 256, 0, stream>>>(
            H(A_S2), H(A_S0), IN(T0 + 9), IN(T0 + 8), nilf, nilf, 0, 32, 288, 288, 25, 192, 1);
        stats(H(A_S0), 2, 192, N1); bnfin(2, IN(T0 + 1), IN(T0 + 0), inv1, 192);
        LGEMM(192, true, 1, 0, H(A_S0), H(OFF_PW1_0), IN(T0 + 12), H(A_S1), N1, 96, 25, 288, 288, 1, 0, SC(2), SH(2));
        stats(H(A_S1), 3, 96, N1); bnfin(3, IN(T0 + 3), IN(T0 + 2), inv1, 96);
        LGEMM( 96, true, 0, 0, H(A_S1), H(OFF_PW2_0), IN(T0 + 14), H(A_S0), N1, 192, 25, 288, 288, 1, 0, SC(3), SH(3));
        stats(H(A_S0), 4, 192, N1); bnfin(4, IN(T0 + 5), IN(T0 + 4), inv1, 192);
        long long tot2 = N2 * 192;
        k_dwconv<<<(int)((tot2 + 255) / 256), 256, 0, stream>>>(
            H(A_S0), H(A_S1), IN(T0 + 11), IN(T0 + 10), SC(4), SH(4), 1, 32, 288, 144, 25, 192, 2);
        stats(H(A_S1), 5, 192, N2); bnfin(5, IN(T0 + 7), IN(T0 + 6), inv2, 192);
        LGEMM(192, false, 0, 0, H(A_S2), H(OFF_RESW), IN(T0 + 16), H(A_S3), N2, 192, 25, 144, 288, 2, 0, nilf, nilf);
        stats(H(A_S3), 6, 192, N2); bnfin(6, IN(T0 + 18), IN(T0 + 17), inv2, 192);
        k_comb_tcn<<<(int)((tot2 + 255) / 256), 256, 0, stream>>>(
            H(A_S1), H(A_S3), SC(5), SH(5), SC(6), SH(6), 1, H(A_S0), tot2, 192);  // x2 @ S0
    }

    // 5. TCN layer 1 (stride 1), input x2 @ S0
    {
        long long tot2 = N2 * 192;
        k_dwconv<<<(int)((tot2 + 255) / 256), 256, 0, stream>>>(
            H(A_S0), H(A_S1), IN(T1 + 9), IN(T1 + 8), nilf, nilf, 0, 32, 144, 144, 25, 192, 1);
        stats(H(A_S1), 7, 192, N2); bnfin(7, IN(T1 + 1), IN(T1 + 0), inv2, 192);
        LGEMM(192, true, 1, 0, H(A_S1), H(OFF_PW1_1), IN(T1 + 12), H(A_S2), N2, 96, 25, 144, 144, 1, 0, SC(7), SH(7));
        stats(H(A_S2), 8, 96, N2); bnfin(8, IN(T1 + 3), IN(T1 + 2), inv2, 96);
        LGEMM( 96, true, 0, 0, H(A_S2), H(OFF_PW2_1), IN(T1 + 14), H(A_S1), N2, 192, 25, 144, 144, 1, 0, SC(8), SH(8));
        stats(H(A_S1), 9, 192, N2); bnfin(9, IN(T1 + 5), IN(T1 + 4), inv2, 192);
        k_dwconv<<<(int)((tot2 + 255) / 256), 256, 0, stream>>>(
            H(A_S1), H(A_S2), IN(T1 + 11), IN(T1 + 10), SC(9), SH(9), 1, 32, 144, 144, 25, 192, 1);
        stats(H(A_S2), 10, 192, N2); bnfin(10, IN(T1 + 7), IN(T1 + 6), inv2, 192);
        k_comb_tcn<<<(int)((tot2 + 255) / 256), 256, 0, stream>>>(
            H(A_S2), H(A_S0), SC(10), SH(10), nilf, nilf, 0, H(A_S1), tot2, 192);  // x3 @ S1
    }

    // 6. ST joint attention on x3 @ S1
    {
        k_pool<<<32 * 169, 192, 0, stream>>>(H(A_S1), H(A_P), 32, 144, 25, 192);
        LGEMM(192, false, 0, 0, H(A_P), H(OFF_FCNW), IN(I_ATT_FCN_B), H(A_FC), 5408LL, 96, 1, 169, 169, 1, 0, nilf, nilf);
        stats(H(A_FC), 11, 96, 5408);
        bnfin(11, IN(I_ATT_FCN_BN_G), IN(I_ATT_FCN_BN_B), invP, 96);
        // a_t: rows (b,t) from pooled[b*169+t]; a_v: rows (b,v) from pooled[b*169+144+v]
        LGEMM(96, true, 2, 1, H(A_FC), H(OFF_CTW), IN(I_ATT_CT_B), H(A_AT), 4608LL, 192, 1, 144, 169, 1,   0, SC(11), SH(11));
        LGEMM(96, true, 2, 1, H(A_FC), H(OFF_CVW), IN(I_ATT_CV_B), H(A_AV),  800LL, 192, 1,  25, 169, 1, 144, SC(11), SH(11));
        long long tot2 = N2 * 192;
        k_mul<<<(int)((tot2 + 255) / 256), 256, 0, stream>>>(
            H(A_S1), H(A_AT), H(A_AV), H(A_S0), 32, 144, 25, 192);                  // m_raw @ S0
        stats(H(A_S0), 12, 192, N2); bnfin(12, IN(I_ATT_BN_G), IN(I_ATT_BN_B), inv2, 192);
        k_final<<<(int)((tot2 + 255) / 256), 256, 0, stream>>>(
            H(A_S0), H(A_S1), SC(12), SH(12), (float*)d_out, 32, 144, 25, 192);
    }
}
